// SAGE_29051158790701
// MI455X (gfx1250) — matrix operand order repaired and checked by reading, not compiled, not run on hardware
//
#include <hip/hip_runtime.h>
#include <hip/hip_bf16.h>
#include <math.h>

// ---------------------------------------------------------------------------
// Types for CDNA5 WMMA (gfx1250, wave32)
// ---------------------------------------------------------------------------
typedef __attribute__((ext_vector_type(16))) __bf16          v16bf;
typedef __attribute__((ext_vector_type(16))) unsigned short  v16us;
typedef __attribute__((ext_vector_type(8)))  unsigned short  v8us;
typedef __attribute__((ext_vector_type(8)))  float           v8f;

__device__ __forceinline__ unsigned short f32_to_bf16(float f) {
  unsigned int u = __float_as_uint(f);
  unsigned int r = u + 0x7fffu + ((u >> 16) & 1u);   // round-to-nearest-even
  return (unsigned short)(r >> 16);
}

__device__ __forceinline__ void atomic_max_f32(float* addr, float val) {
  int* ai = (int*)addr;
  int cur = __float_as_int(*addr);
  while (__int_as_float(cur) < val) {
    int old = atomicCAS(ai, cur, __float_as_int(val));
    if (old == cur) break;
    cur = old;
  }
}

// ---------------------------------------------------------------------------
// GEMM: C[N,Ncols] = A[N,256](bf16) * B[256,Npad](bf16, fragment-packed)
// One wave per 16x64 output tile: 1 A fragment x 4 B fragments per k-step,
// K=256 fixed, fully unrolled -> 32 v_wmma_f32_16x16x32_bf16 per wave.
// Bpack layout: [kt][nt][lane][e] (16 contiguous bf16 per lane per fragment).
// ---------------------------------------------------------------------------
__global__ __launch_bounds__(32)
void gat_gemm_wmma(const unsigned short* __restrict__ Abf,
                   const unsigned short* __restrict__ Bpack,
                   float* __restrict__ C,
                   int N, int NT /* Npad/16 */, int Ncols) {
  const int K = 256;
  const int lane = threadIdx.x & 31;
  const int r16  = lane & 15;        // row-in-tile (A/M), col-in-tile (B,C/N)
  const int half = lane >> 4;        // K sub-range selector

  const long m0  = (long)blockIdx.x * 16;
  const int  nt0 = blockIdx.y * 4;

  v8f acc0 = {}, acc1 = {}, acc2 = {}, acc3 = {};

  // A: row = m0+r16; per k-tile need K = k0 + half*8 + {0..7} and +16
  const unsigned short* arow = Abf + (m0 + r16) * (long)K + half * 8;

#pragma unroll
  for (int kt = 0; kt < 8; ++kt) {   // 8 k-tiles of 32
    union { v16us v; v8us h[2]; } aw;
    aw.h[0] = *(const v8us*)(arow + kt * 32);        // elems 0..7
    aw.h[1] = *(const v8us*)(arow + kt * 32 + 16);   // elems 8..15
    v16bf a;
    __builtin_memcpy(&a, &aw.v, sizeof(a));

    const unsigned short* bbase =
        Bpack + (((long)kt * NT + nt0) * 32 + lane) * 16;
    v16us bu0 = *(const v16us*)(bbase);
    v16us bu1 = *(const v16us*)(bbase + 512);        // next n-tile: 32*16
    v16us bu2 = *(const v16us*)(bbase + 1024);
    v16us bu3 = *(const v16us*)(bbase + 1536);
    v16bf b0, b1, b2, b3;
    __builtin_memcpy(&b0, &bu0, sizeof(b0));
    __builtin_memcpy(&b1, &bu1, sizeof(b1));
    __builtin_memcpy(&b2, &bu2, sizeof(b2));
    __builtin_memcpy(&b3, &bu3, sizeof(b3));

    acc0 = __builtin_amdgcn_wmma_f32_16x16x32_bf16(false, a, false, b0, (short)0, acc0, false, false);
    acc1 = __builtin_amdgcn_wmma_f32_16x16x32_bf16(false, a, false, b1, (short)0, acc1, false, false);
    acc2 = __builtin_amdgcn_wmma_f32_16x16x32_bf16(false, a, false, b2, (short)0, acc2, false, false);
    acc3 = __builtin_amdgcn_wmma_f32_16x16x32_bf16(false, a, false, b3, (short)0, acc3, false, false);
  }

  // D layout: VGPR r -> row = r + 8*half, col = lane&15 within tile
  v8f accs[4] = {acc0, acc1, acc2, acc3};
#pragma unroll
  for (int j = 0; j < 4; ++j) {
    const int col = (nt0 + j) * 16 + r16;
    if (col < Ncols) {
#pragma unroll
      for (int r = 0; r < 8; ++r) {
        long row = m0 + r + half * 8;
        C[row * (long)Ncols + col] = accs[j][r];
      }
    }
  }
}

// ---------------------------------------------------------------------------
// Activations f32 -> bf16 row-major (2 elems/thread)
// ---------------------------------------------------------------------------
__global__ void gat_convert_a(const float* __restrict__ in,
                              unsigned int* __restrict__ out, long n2) {
  long i = (long)blockIdx.x * blockDim.x + threadIdx.x;
  if (i >= n2) return;
  float2 v = ((const float2*)in)[i];
  out[i] = (unsigned int)f32_to_bf16(v.x) | ((unsigned int)f32_to_bf16(v.y) << 16);
}

// ---------------------------------------------------------------------------
// Weights [256, Nsrc] f32 -> fragment-packed bf16 [kt][nt][lane][e]
// element e of lane L in tile (kt, nt) = B[kt*32 + (e>>3)*16 + (L>>4)*8 + (e&7)][nt*16 + (L&15)]
// ---------------------------------------------------------------------------
__global__ void gat_pack_w(const float* __restrict__ W,
                           unsigned short* __restrict__ out,
                           int NT, int Nsrc) {
  int o = blockIdx.x * blockDim.x + threadIdx.x;
  int total = 8 * NT * 32 * 16;                  // K=256 -> 8 k-tiles
  if (o >= total) return;
  int e    = o & 15;
  int L    = (o >> 4) & 31;
  int rest = o >> 9;
  int nt   = rest % NT;
  int kt   = rest / NT;
  int k = kt * 32 + (e >> 3) * 16 + (L >> 4) * 8 + (e & 7);
  int n = nt * 16 + (L & 15);
  out[o] = (n < Nsrc) ? f32_to_bf16(W[(long)k * Nsrc + n]) : (unsigned short)0;
}

// el[n,h] = sum_d feat[n,h,d]*al[h,d] ; er likewise
__global__ void gat_attn_proj(const float* __restrict__ feat,
                              const float* __restrict__ al,
                              const float* __restrict__ ar,
                              float* __restrict__ el, float* __restrict__ er,
                              int N, int H, int D) {
  int idx = blockIdx.x * blockDim.x + threadIdx.x;
  if (idx >= N * H) return;
  int n = idx / H, h = idx - n * H;
  const float* f = feat + (long)n * H * D + (long)h * D;
  const float* a = al + (long)h * D;
  const float* b = ar + (long)h * D;
  float sl = 0.f, sr = 0.f;
  for (int d = 0; d < D; ++d) { float x = f[d]; sl += x * a[d]; sr += x * b[d]; }
  el[idx] = sl;
  er[idx] = sr;
}

__global__ void gat_fill(float* __restrict__ p, float v, int n) {
  int idx = blockIdx.x * blockDim.x + threadIdx.x;
  if (idx < n) p[idx] = v;
}

__global__ void gat_init_bias(float* __restrict__ out, const float* __restrict__ bias,
                              long total, int F) {
  long idx = (long)blockIdx.x * blockDim.x + threadIdx.x;
  if (idx < total) out[idx] = bias[idx % F];
}

__device__ __forceinline__ float gat_logit(const float* el, const float* er,
                                           int s, int d, int H, int h) {
  float v = el[(long)s * H + h] + er[(long)d * H + h];
  return v > 0.f ? v : 0.2f * v;   // leaky_relu, slope 0.2
}

__global__ void gat_edge_max(const int* __restrict__ src, const int* __restrict__ dst,
                             const float* __restrict__ el, const float* __restrict__ er,
                             float* __restrict__ mx, int E, int H) {
  int idx = blockIdx.x * blockDim.x + threadIdx.x;
  if (idx >= E * H) return;
  int e = idx / H, h = idx - e * H;
  int s = src[e], d = dst[e];
  atomic_max_f32(&mx[(long)d * H + h], gat_logit(el, er, s, d, H, h));
}

__global__ void gat_edge_expsum(const int* __restrict__ src, const int* __restrict__ dst,
                                const float* __restrict__ el, const float* __restrict__ er,
                                const float* __restrict__ mx, float* __restrict__ dn,
                                int E, int H) {
  int idx = blockIdx.x * blockDim.x + threadIdx.x;
  if (idx >= E * H) return;
  int e = idx / H, h = idx - e * H;
  int s = src[e], d = dst[e];
  float ex = __expf(gat_logit(el, er, s, d, H, h) - mx[(long)d * H + h]);
  atomicAdd(&dn[(long)d * H + h], ex);
}

// One wave per edge: lane strides over the H*D feature columns.
__global__ void gat_edge_agg(const int* __restrict__ src, const int* __restrict__ dst,
                             const float* __restrict__ el, const float* __restrict__ er,
                             const float* __restrict__ mx, const float* __restrict__ dn,
                             const float* __restrict__ feat, float* __restrict__ out,
                             int E, int H, int D) {
  int wave = (blockIdx.x * blockDim.x + threadIdx.x) >> 5;
  int lane = threadIdx.x & 31;
  if (wave >= E) return;
  int s = src[wave], d = dst[wave];
  int F = H * D;
  for (int f = lane; f < F; f += 32) {
    int h = f / D;
    float e = gat_logit(el, er, s, d, H, h);
    float alpha = __expf(e - mx[(long)d * H + h]) / dn[(long)d * H + h];
    atomicAdd(&out[(long)d * F + f], feat[(long)s * F + f] * alpha);
  }
}

// ---------------------------------------------------------------------------
// Launch
// ---------------------------------------------------------------------------
extern "C" void kernel_launch(void* const* d_in, const int* in_sizes, int n_in,
                              void* d_out, int out_size, void* d_ws, size_t ws_size,
                              hipStream_t stream) {
  (void)n_in; (void)out_size; (void)ws_size;
  const float* x   = (const float*)d_in[0];
  const int*   src = (const int*)d_in[1];
  const int*   dst = (const int*)d_in[2];
  const float* W1  = (const float*)d_in[3];
  const float* al1 = (const float*)d_in[4];
  const float* ar1 = (const float*)d_in[5];
  const float* b1  = (const float*)d_in[6];
  const float* W2  = (const float*)d_in[7];
  const float* al2 = (const float*)d_in[8];
  const float* ar2 = (const float*)d_in[9];
  const float* b2  = (const float*)d_in[10];
  const float* W3  = (const float*)d_in[11];
  const float* al3 = (const float*)d_in[12];
  const float* ar3 = (const float*)d_in[13];
  const float* b3  = (const float*)d_in[14];

  const int IN = 256, H = 4, D = 64, F = H * D;   // F == 256
  const int N = in_sizes[0] / IN;                 // 50000 (multiple of 16)
  const int E = in_sizes[1];                      // 800000

  // Workspace carve-up (256B aligned regions)
  char* wsp = (char*)d_ws;
  auto carve = [&](size_t bytes) {
    void* p = (void*)wsp;
    wsp += (bytes + 255) & ~(size_t)255;
    return p;
  };
  float* featA = (float*)carve((size_t)N * F * sizeof(float));
  float* hbuf  = (float*)carve((size_t)N * F * sizeof(float));
  unsigned short* Abf = (unsigned short*)carve((size_t)N * F * sizeof(unsigned short));
  float* el    = (float*)carve((size_t)N * H * sizeof(float));
  float* er    = (float*)carve((size_t)N * H * sizeof(float));
  float* mx    = (float*)carve((size_t)N * H * sizeof(float));
  float* dn    = (float*)carve((size_t)N * H * sizeof(float));
  unsigned short* Wpack = (unsigned short*)carve((size_t)IN * 256 * sizeof(unsigned short));

  auto layer = [&](const float* hin, const float* W, const float* alp,
                   const float* arp, const float* bias,
                   int h, int dd, int nsrc, int npad, float* outp) {
    const int f  = h * dd;                        // == nsrc
    const int nt = npad / 16;
    // 1) activations -> bf16 row-major
    long n2 = (long)N * IN / 2;
    gat_convert_a<<<(unsigned)((n2 + 255) / 256), 256, 0, stream>>>(
        hin, (unsigned int*)Abf, n2);
    // 2) weights -> bf16 fragment-packed (zero pad columns >= nsrc)
    int wtot = IN * npad;
    gat_pack_w<<<(wtot + 255) / 256, 256, 0, stream>>>(W, Wpack, nt, nsrc);
    // 3) dense projection via WMMA (16x64 tile per wave)
    dim3 g(N / 16, npad / 64);
    gat_gemm_wmma<<<g, 32, 0, stream>>>(Abf, Wpack, featA, N, nt, nsrc);
    // 4) per-node attention logits
    int nh = N * h;
    gat_attn_proj<<<(nh + 255) / 256, 256, 0, stream>>>(featA, alp, arp, el, er, N, h, dd);
    // 5) segment softmax init
    gat_fill<<<(nh + 255) / 256, 256, 0, stream>>>(mx, -INFINITY, nh);
    gat_fill<<<(nh + 255) / 256, 256, 0, stream>>>(dn, 0.f, nh);
    // 6) segment max / exp-sum
    int eh = E * h;
    gat_edge_max<<<(eh + 255) / 256, 256, 0, stream>>>(src, dst, el, er, mx, E, h);
    gat_edge_expsum<<<(eh + 255) / 256, 256, 0, stream>>>(src, dst, el, er, mx, dn, E, h);
    // 7) out = bias, then alpha-weighted scatter-add of feat[src]
    long tot = (long)N * f;
    gat_init_bias<<<(unsigned)((tot + 255) / 256), 256, 0, stream>>>(outp, bias, tot, f);
    gat_edge_agg<<<(E + 7) / 8, 256, 0, stream>>>(src, dst, el, er, mx, dn, featA, outp, E, h, dd);
  };

  // Layer 1: x[N,256] -> hbuf[N,256]
  layer(x, W1, al1, ar1, b1, H, D, F, F, hbuf);
  // Layer 2: hbuf -> hbuf (input consumed by convert/GEMM before overwrite)
  layer(hbuf, W2, al2, ar2, b2, H, D, F, F, hbuf);
  // Layer 3: hbuf -> d_out [N,40]; heads=1 so mean(axis=1) is identity
  layer(hbuf, W3, al3, ar3, b3, 1, 40, 40, 64, (float*)d_out);
}
